// GraphSAGE_31018253812109
// MI455X (gfx1250) — matrix operand order repaired and checked by reading, not compiled, not run on hardware
//
#include <hip/hip_runtime.h>
#include <hip/hip_bf16.h>

// ---------------------------------------------------------------------------
// GraphSAGE link prediction for MI455X (gfx1250, wave32, WMMA)
// N=100000 nodes, D=128, E=800000 edges, P=100000 pos/neg pairs.
// Dense D x D GEMMs run on v_wmma_f32_16x16x32_bf16 (bf16 in, f32 accum),
// weights staged into LDS with global_load_async_to_lds_b128 (ASYNCcnt).
// Edge aggregation is the bandwidth-critical part -> coalesced
// one-wave-per-edge scatter with native global_atomic_add_f32
// (unsafeAtomicAdd; msum is 51MB and stays resident in the 192MB L2).
// ---------------------------------------------------------------------------

constexpr int DIM = 128;
constexpr int PW_ELEMS = 16384;  // packed weight: 4 kchunks * 8 ntiles * 32 lanes * 16 elems

typedef __bf16 v16bf __attribute__((ext_vector_type(16)));
typedef __bf16 v8bf  __attribute__((ext_vector_type(8)));
typedef float  v8f   __attribute__((ext_vector_type(8)));

__device__ __forceinline__ unsigned short f2bf(float f) {
  unsigned int u = __float_as_uint(f);
  u += 0x7FFFu + ((u >> 16) & 1u);          // round-to-nearest-even
  return (unsigned short)(u >> 16);
}
__device__ __forceinline__ float bf2f(unsigned short h) {
  return __uint_as_float(((unsigned int)h) << 16);
}

// ---------------------------------------------------------------------------
__global__ __launch_bounds__(256) void zero_f32(float* __restrict__ p, long n) {
  long i = (long)blockIdx.x * blockDim.x + threadIdx.x;
  if (i < n) p[i] = 0.0f;
}

__global__ __launch_bounds__(256) void deg_acc(const int* __restrict__ dst,
                                               float* __restrict__ deg, long E) {
  long i = (long)blockIdx.x * blockDim.x + threadIdx.x;
  if (i < E) unsafeAtomicAdd(&deg[dst[i]], 1.0f);   // native global_atomic_add_f32
}

__global__ __launch_bounds__(256) void inv_deg(float* __restrict__ deg, long n) {
  long i = (long)blockIdx.x * blockDim.x + threadIdx.x;
  if (i < n) deg[i] = 1.0f / fmaxf(deg[i], 1.0f);
}

// One wave per edge: 32 lanes x float4 = full 128-wide row. Coalesced gather,
// row-contiguous f32 atomic scatter (lands in L2; msum is 51MB < 192MB L2).
// unsafeAtomicAdd forces the hardware no-return global_atomic_add_f32 instead
// of a CAS-loop expansion.
__global__ __launch_bounds__(256) void scatter_rows(const float* __restrict__ h,
                                                    const int* __restrict__ src,
                                                    const int* __restrict__ dst,
                                                    float* __restrict__ msum, long E) {
  long tid = (long)blockIdx.x * blockDim.x + threadIdx.x;
  long e = tid >> 5;
  if (e >= E) return;
  int c = (int)(tid & 31);
  const float4 v = *(const float4*)(h + (long)src[e] * DIM + c * 4);
  float* p = msum + (long)dst[e] * DIM + c * 4;
  unsafeAtomicAdd(p + 0, v.x);
  unsafeAtomicAdd(p + 1, v.y);
  unsafeAtomicAdd(p + 2, v.z);
  unsafeAtomicAdd(p + 3, v.w);
}

// fp32 h / (msum * 1/deg) -> bf16 copies for the WMMA GEMM
__global__ __launch_bounds__(256) void convert_layer(const float* __restrict__ h,
                                                     const float* __restrict__ msum,
                                                     const float* __restrict__ invdeg,
                                                     unsigned short* __restrict__ hb,
                                                     unsigned short* __restrict__ hnb,
                                                     long total) {
  long i = (long)blockIdx.x * blockDim.x + threadIdx.x;
  if (i >= total) return;
  hb[i] = f2bf(h[i]);
  hnb[i] = f2bf(msum[i] * invdeg[i >> 7]);
}

// Pack one fp32 [128,128] row-major weight into per-lane-contiguous WMMA B
// fragments. Contiguous-k B layout (this code's choice) for 32x16 tile: element e of the per-lane
// 16-elem vector holds K = (lane/16)*16 + e, column N = ntile*16 + lane%16.
__global__ __launch_bounds__(256) void pack_weights(const float* __restrict__ W,
                                                    unsigned short* __restrict__ out) {
  int i = blockIdx.x * 256 + threadIdx.x;   // [0, PW_ELEMS)
  int e = i & 15;
  int lane = (i >> 4) & 31;
  int tile = i >> 9;                         // kc*8 + nt
  int kc = tile >> 3, nt = tile & 7;
  int hl = lane >> 4, r = lane & 15;
  int k = kc * 32 + hl * 16 + e;
  int n = nt * 16 + r;
  out[i] = f2bf(W[k * DIM + n]);
}

// z = h[s] * h[d] elementwise -> bf16
__global__ __launch_bounds__(256) void zmul(const float* __restrict__ h,
                                            const int* __restrict__ s,
                                            const int* __restrict__ d,
                                            unsigned short* __restrict__ zb, long total) {
  long i = (long)blockIdx.x * blockDim.x + threadIdx.x;
  if (i >= total) return;
  long row = i >> 7;
  int c = (int)(i & 127);
  zb[i] = f2bf(h[(long)s[row] * DIM + c] * h[(long)d[row] * DIM + c]);
}

// ---------------------------------------------------------------------------
// WMMA GEMM: out[M,128] = A1@W1 (+ A2@W2) + bias, optional ReLU.
// Block = 256 threads = 8 waves; each wave owns one 16-row x 128-col tile.
// Packed weights staged straight into LDS with GLOBAL_LOAD_ASYNC_TO_LDS_B128
// (GVS addressing: SGPR-pair base + per-lane 32-bit byte offset; VDST is the
// per-lane LDS byte address), synchronized with s_wait_asynccnt 0 + barrier.
// A fragment (contiguous-k 16x32, same k order as packed B): lane (hl,r) holds row m0+r,
// elems 0..15 = K in [kc*32+hl*16, +16).
// ---------------------------------------------------------------------------
__global__ __launch_bounds__(256)
void gemm_bf16(const unsigned short* __restrict__ A1,
               const unsigned short* __restrict__ A2,
               const unsigned short* __restrict__ W1p,
               const unsigned short* __restrict__ W2p,
               const float* __restrict__ bias,
               float* __restrict__ outF,
               unsigned short* __restrict__ outB,
               int M, int relu) {
  __shared__ unsigned short sW[2 * PW_ELEMS];
  const int t = threadIdx.x;
  const bool dual = (A2 != nullptr);
  const unsigned sBase = (unsigned)(uintptr_t)&sW[0];   // LDS segment byte offset
  // async copy: 2048 (or 4096) x 16B chunks, 8 (16) per thread
  {
    const int nChunks = dual ? 4096 : 2048;
    for (int i = t; i < nChunks; i += 256) {
      const unsigned ldsAddr = sBase + (unsigned)i * 16u;
      if (i < 2048) {
        const unsigned gOff = (unsigned)i * 16u;
        asm volatile("global_load_async_to_lds_b128 %0, %1, %2"
                     :: "v"(ldsAddr), "v"(gOff), "s"(W1p) : "memory");
      } else {
        const unsigned gOff = (unsigned)(i - 2048) * 16u;
        asm volatile("global_load_async_to_lds_b128 %0, %1, %2"
                     :: "v"(ldsAddr), "v"(gOff), "s"(W2p) : "memory");
      }
    }
    asm volatile("s_wait_asynccnt 0x0" ::: "memory");
  }
  __syncthreads();

  const int lane = t & 31;
  const int hl = lane >> 4;
  const int r = lane & 15;
  const long m0 = ((long)blockIdx.x * 8 + (t >> 5)) * 16;
  if (m0 >= M) return;                           // wave-uniform (EXEC stays all-ones)

  v16bf a1[4], a2[4];
  const unsigned short* row1 = A1 + (m0 + r) * DIM;
#pragma unroll
  for (int kc = 0; kc < 4; ++kc) {
    const int kb = kc * 32 + hl * 16;
    v8bf lo = *(const v8bf*)(row1 + kb);
    v8bf hi = *(const v8bf*)(row1 + kb + 8);
    a1[kc] = __builtin_shufflevector(lo, hi, 0, 1, 2, 3, 4, 5, 6, 7,
                                     8, 9, 10, 11, 12, 13, 14, 15);
  }
  if (dual) {
    const unsigned short* row2 = A2 + (m0 + r) * DIM;
#pragma unroll
    for (int kc = 0; kc < 4; ++kc) {
      const int kb = kc * 32 + hl * 16;
      v8bf lo = *(const v8bf*)(row2 + kb);
      v8bf hi = *(const v8bf*)(row2 + kb + 8);
      a2[kc] = __builtin_shufflevector(lo, hi, 0, 1, 2, 3, 4, 5, 6, 7,
                                       8, 9, 10, 11, 12, 13, 14, 15);
    }
  }

#pragma unroll
  for (int nt = 0; nt < 8; ++nt) {
    v8f acc = {};
    if (dual) {
#pragma unroll
      for (int kc = 0; kc < 4; ++kc) {
        v16bf b1 = *(const v16bf*)(sW + (((kc * 8 + nt) * 32) + lane) * 16);
        acc = __builtin_amdgcn_wmma_f32_16x16x32_bf16(false, a1[kc], false, b1,
                                                      (short)0, acc, false, false);
        v16bf b2 = *(const v16bf*)(sW + PW_ELEMS + (((kc * 8 + nt) * 32) + lane) * 16);
        acc = __builtin_amdgcn_wmma_f32_16x16x32_bf16(false, a2[kc], false, b2,
                                                      (short)0, acc, false, false);
      }
    } else {
#pragma unroll
      for (int kc = 0; kc < 4; ++kc) {
        v16bf b1 = *(const v16bf*)(sW + (((kc * 8 + nt) * 32) + lane) * 16);
        acc = __builtin_amdgcn_wmma_f32_16x16x32_bf16(false, a1[kc], false, b1,
                                                      (short)0, acc, false, false);
      }
    }
    const int n = nt * 16 + r;
    const float bv = bias ? bias[n] : 0.0f;
    // C/D layout: VGPR v, lane (hl,r) -> row m0 + hl*8 + v, col n
#pragma unroll
    for (int v = 0; v < 8; ++v) {
      float o = acc[v] + bv;
      if (relu) o = fmaxf(o, 0.0f);
      const long row = m0 + hl * 8 + v;
      if (outF) outF[row * DIM + n] = o;
      if (outB) outB[row * DIM + n] = f2bf(o);
    }
  }
}

// 128 -> 2 logits + softmax[:,1]
__global__ __launch_bounds__(256) void final_pred(const unsigned short* __restrict__ t2b,
                                                  const float* __restrict__ Wp2,
                                                  const float* __restrict__ bp2,
                                                  float* __restrict__ out, long P) {
  long i = (long)blockIdx.x * blockDim.x + threadIdx.x;
  if (i >= P) return;
  float l0 = bp2[0], l1 = bp2[1];
  const unsigned short* row = t2b + i * DIM;
#pragma unroll 4
  for (int k = 0; k < DIM; ++k) {
    float z = bf2f(row[k]);
    l0 = fmaf(z, Wp2[2 * k + 0], l0);
    l1 = fmaf(z, Wp2[2 * k + 1], l1);
  }
  float m = fmaxf(l0, l1);
  float e0 = __expf(l0 - m), e1 = __expf(l1 - m);
  out[i] = e1 / (e0 + e1);
}

// ---------------------------------------------------------------------------
extern "C" void kernel_launch(void* const* d_in, const int* in_sizes, int n_in,
                              void* d_out, int out_size, void* d_ws, size_t ws_size,
                              hipStream_t stream) {
  const float* x   = (const float*)d_in[0];
  const int* src   = (const int*)d_in[1];
  const int* dst   = (const int*)d_in[2];
  const int* psel[2][2] = {{(const int*)d_in[3], (const int*)d_in[4]},
                           {(const int*)d_in[5], (const int*)d_in[6]}};
  const float* Ws[3] = {(const float*)d_in[7], (const float*)d_in[10], (const float*)d_in[13]};
  const float* Wn[3] = {(const float*)d_in[8], (const float*)d_in[11], (const float*)d_in[14]};
  const float* bL[3] = {(const float*)d_in[9], (const float*)d_in[12], (const float*)d_in[15]};
  const float* Wp0 = (const float*)d_in[16];
  const float* bp0 = (const float*)d_in[17];
  const float* Wp1 = (const float*)d_in[18];
  const float* bp1 = (const float*)d_in[19];
  const float* Wp2 = (const float*)d_in[20];
  const float* bp2 = (const float*)d_in[21];

  const long N = in_sizes[0] / DIM;
  const long E = in_sizes[1];
  const long P = in_sizes[3];

  // workspace carve-out (~155 MB)
  char* w = (char*)d_ws;
  size_t off = 0;
  auto take = [&](size_t bytes) -> void* {
    void* p = w + off;
    off += (bytes + 511) & ~(size_t)511;
    return p;
  };
  float* hcur = (float*)take((size_t)N * DIM * 4);           // node features (ping)
  float* msum = (float*)take((size_t)N * DIM * 4);           // neighbor sum
  float* degp = (float*)take((size_t)N * 4);                 // degree -> 1/deg
  unsigned short* hb  = (unsigned short*)take((size_t)N * DIM * 2);  // bf16 A buffers
  unsigned short* hnb = (unsigned short*)take((size_t)N * DIM * 2);
  unsigned short* pw  = (unsigned short*)take((size_t)8 * PW_ELEMS * 2);

  // Pack all D x D weights into WMMA B-fragment layout (bf16)
  const float* wsrc[8] = {Ws[0], Wn[0], Ws[1], Wn[1], Ws[2], Wn[2], Wp0, Wp1};
  for (int m = 0; m < 8; ++m)
    pack_weights<<<PW_ELEMS / 256, 256, 0, stream>>>(wsrc[m], pw + (size_t)m * PW_ELEMS);

  // Degrees (once; graph fixed across layers)
  zero_f32<<<(int)((N + 255) / 256), 256, 0, stream>>>(degp, N);
  deg_acc<<<(int)((E + 255) / 256), 256, 0, stream>>>(dst, degp, E);
  inv_deg<<<(int)((N + 255) / 256), 256, 0, stream>>>(degp, N);

  const long ND = N * DIM;
  const int gemmBlocksN = (int)((N / 16 + 7) / 8);
  const float* hin = x;
  for (int L = 0; L < 3; ++L) {
    zero_f32<<<(int)((ND + 255) / 256), 256, 0, stream>>>(msum, ND);
    scatter_rows<<<(int)((E * 32 + 255) / 256), 256, 0, stream>>>(hin, src, dst, msum, E);
    convert_layer<<<(int)((ND + 255) / 256), 256, 0, stream>>>(hin, msum, degp, hb, hnb, ND);
    gemm_bf16<<<gemmBlocksN, 256, 0, stream>>>(
        hb, hnb, pw + (size_t)(2 * L) * PW_ELEMS, pw + (size_t)(2 * L + 1) * PW_ELEMS,
        bL[L], hcur, nullptr, (int)N, (L < 2) ? 1 : 0);
    hin = hcur;   // in-place ping: gemm inputs (hb/hnb) already consumed hin
  }

  // Predictor: z -> relu(z@Wp0+bp0) -> relu(@Wp1+bp1) -> 128->2 softmax[:,1]
  const long PD = P * DIM;
  const int gemmBlocksP = (int)((P / 16 + 7) / 8);
  for (int s = 0; s < 2; ++s) {
    zmul<<<(int)((PD + 255) / 256), 256, 0, stream>>>(hcur, psel[s][0], psel[s][1], hb, PD);
    gemm_bf16<<<gemmBlocksP, 256, 0, stream>>>(hb, nullptr, pw + (size_t)6 * PW_ELEMS, nullptr,
                                               bp0, nullptr, hnb, (int)P, 1);
    gemm_bf16<<<gemmBlocksP, 256, 0, stream>>>(hnb, nullptr, pw + (size_t)7 * PW_ELEMS, nullptr,
                                               bp1, nullptr, hb, (int)P, 1);
    final_pred<<<(int)((P + 255) / 256), 256, 0, stream>>>(hb, Wp2, bp2,
                                                           (float*)d_out + (size_t)s * P, P);
  }
  (void)n_in; (void)out_size; (void)ws_size;
}
